// Spectral_Discriminator_893353198109
// MI455X (gfx1250) — matrix operand order repaired and checked by reading, not compiled, not run on hardware
//
#include <hip/hip_runtime.h>
#include <hip/hip_bf16.h>
#include <stdint.h>

// ---------------------------------------------------------------------------
// Spectral discriminator on MI455X (gfx1250, wave32, WMMA).
// gray-fold -> 2D DCT (2 GEMMs) -> 2D FFT (fused real GEMMs) -> log|F| ->
// radial profile -> minmax norm -> 90-tap head.
// bf16-in / f32-acc v_wmma_f32_16x16x32_bf16; producer-side fragment packing;
// 4-image batching per wave to amortize shared DCT/DFT basis fragments.
// ---------------------------------------------------------------------------

typedef __attribute__((ext_vector_type(16))) __bf16 bf16x16;
typedef __attribute__((ext_vector_type(4)))  __bf16 bf16x4;
typedef __attribute__((ext_vector_type(8)))  float  f32x8;

#define PI_OVER_512 0.006135923151542565f
#define PI_OVER_128 0.024543692606170260f
#define IMGS 4   // images batched per wave in GEMM kernels

// A-pack: 256x256 matrix as 16x8 grid of 16(M)x32(K) tiles, each tile 512 bf16,
//   element (lane L, e): m = L&15, kk = (e>>3)*16 + (L>>4)*8 + (e&7)
// B-pack: 256x256 matrix as 8x16 grid of 32(K)x16(N) tiles, each tile 512 bf16,
//   element (lane L, e): kk = (e>>3)*16 + (L>>4)*8 + (e&7), n = L&15

__device__ __forceinline__ f32x8 wmma_bf16(bf16x16 a, bf16x16 b, f32x8 c) {
  return __builtin_amdgcn_wmma_f32_16x16x32_bf16(false, a, false, b, (short)0, c,
                                                 false, false);
}

// scatter offset for writing accumulator element (M,N) into packed layouts
__device__ __forceinline__ int pack_off(int M, int N, int outFmt) {
  if (outFmt == 0) {  // B-pack (output rows become next GEMM's K)
    int bt = ((M >> 5) << 4) | (N >> 4);
    int kk = M & 31, n = N & 15;
    return bt * 512 + ((((kk >> 3) & 1) << 4) | n) * 16 + ((kk & 7) | ((kk >> 4) << 3));
  } else {            // A-pack
    int at = ((M >> 4) << 3) | (N >> 5);
    int kk = N & 31;
    int L2 = (M & 15) | (((kk >> 3) & 1) << 4);
    int e  = (kk & 7) | (((kk >> 4) & 1) << 3);
    return at * 512 + L2 * 16 + e;
  }
}

__device__ __forceinline__ int radial_bin(int row, int col) {
  int sy = (row + 128) & 255;           // fftshift
  int sx = (col + 128) & 255;
  float dy = (float)(sy - 128);
  float dx = (float)(sx - 128);
  return (int)sqrtf(dx * dx + dy * dy); // max 181
}

// -------------------- zero the accumulators (every launch) ------------------
__global__ __launch_bounds__(256) void zero_k(float* nr, float* tbin) {
  int i = blockIdx.x * 256 + threadIdx.x;
  if (i < 192) nr[i] = 0.0f;
  if (i < 128 * 192) tbin[i] = 0.0f;
}

// -------------------- basis matrices, packed, bf16 --------------------------
// order: [0]=CT_B  [1]=C_A  [2]=WR_B  [3]=WI_B  [4]=WR_A  [5]=WI_A
__global__ __launch_bounds__(256) void init_basis(__bf16* basis) {
  int idx   = blockIdx.x * 256 + threadIdx.x;   // 6*65536 total
  int which = idx >> 16;
  int rem   = idx & 65535;
  int tile  = rem >> 9;
  int lane  = (rem >> 4) & 31;
  int e     = rem & 15;
  float val;
  if (which < 2) {  // DCT: C[k,n] = 2 cos(pi*(2n+1)*k/512)
    int tmod;
    if (which == 0) {  // B-pack of C^T: element [w][k'] = C[k'][w]
      int krow = (tile >> 4) * 32 + ((e >> 3) << 4) + ((lane >> 4) << 3) + (e & 7);  // w
      int n    = (tile & 15) * 16 + (lane & 15);             // k'
      tmod = ((2 * krow + 1) * n) & 1023;
    } else {           // A-pack of C
      int m = (tile >> 3) * 16 + (lane & 15);
      int k = (tile & 7) * 32 + ((e >> 3) << 4) + ((lane >> 4) << 3) + (e & 7);
      tmod = ((2 * k + 1) * m) & 1023;
    }
    val = 2.0f * cosf((float)tmod * PI_OVER_512);
  } else {  // DFT basis: Wr = cos(2*pi*j*k/256), Wi = -sin(...)
    int tmod;
    if (which < 4) {   // B-pack
      int krow = (tile >> 4) * 32 + ((e >> 3) << 4) + ((lane >> 4) << 3) + (e & 7);
      int n    = (tile & 15) * 16 + (lane & 15);
      tmod = (krow * n) & 255;
    } else {           // A-pack
      int m = (tile >> 3) * 16 + (lane & 15);
      int k = (tile & 7) * 32 + ((e >> 3) << 4) + ((lane >> 4) << 3) + (e & 7);
      tmod = (m * k) & 255;
    }
    float ang = (float)tmod * PI_OVER_128;
    val = ((which & 1) == 0) ? cosf(ang) : -sinf(ang);
  }
  basis[idx] = (__bf16)val;
}

// -------------------- per-bin pixel counts ----------------------------------
__global__ __launch_bounds__(256) void compute_nr(float* nr) {
  int pix = blockIdx.x * 256 + threadIdx.x;  // 65536 pixels
  atomicAdd(&nr[radial_bin(pix >> 8, pix & 255)], 1.0f);
}

// -------------------- gray fold: f32 HBM read -> bf16 A-pack ----------------
__global__ __launch_bounds__(256) void gray_pack(const float* __restrict__ x,
                                                 __bf16* __restrict__ G) {
  int idx = blockIdx.x * 256 + threadIdx.x;  // 128*16384 quads
  int img = idx >> 14;
  int rem = idx & 16383;
  int h   = rem >> 6;
  int w4  = (rem & 63) << 2;
  const float4* xp = (const float4*)x;
  int64_t base = (int64_t)img * 49152 + h * 64 + (w4 >> 2);
  float4 c0 = xp[base];
  float4 c1 = xp[base + 16384];
  float4 c2 = xp[base + 32768];
  float g0 = 0.2989f * c0.x + 0.587f * c1.x + 0.114f * c2.x;
  float g1 = 0.2989f * c0.y + 0.587f * c1.y + 0.114f * c2.y;
  float g2 = 0.2989f * c0.z + 0.587f * c1.z + 0.114f * c2.z;
  float g3 = 0.2989f * c0.w + 0.587f * c1.w + 0.114f * c2.w;
  int at = ((h >> 4) << 3) | (w4 >> 5);
  int kk = w4 & 31;
  int L  = (h & 15) | (((kk >> 3) & 1) << 4);
  int e0 = (kk & 7) | (((kk >> 4) & 1) << 3);
  bf16x4 v = {(__bf16)g0, (__bf16)g1, (__bf16)g2, (__bf16)g3};
  *(bf16x4*)(G + (int64_t)img * 65536 + at * 512 + L * 16 + e0) = v;
}

// ------------- GEMM, B operand shared (basis), 4 images per wave ------------
__global__ __launch_bounds__(256) void gemm_sB(
    const __bf16* __restrict__ A, int64_t aStride,
    const __bf16* __restrict__ B,
    __bf16* __restrict__ O, int64_t oStride, int outFmt) {
  int wave = threadIdx.x >> 5;
  int lane = threadIdx.x & 31;
  int grp  = blockIdx.x >> 5;                 // image group (4 imgs)
  int tile = ((blockIdx.x & 31) << 3) | wave; // 0..255
  int trow = tile >> 4, tcol = tile & 15;
  int img0 = grp * IMGS;
  f32x8 acc[IMGS] = {};
#pragma unroll
  for (int kt = 0; kt < 8; ++kt) {
    bf16x16 b = *(const bf16x16*)(B + ((kt * 16 + tcol) * 512 + lane * 16));
    int aoff = (trow * 8 + kt) * 512 + lane * 16;
#pragma unroll
    for (int i = 0; i < IMGS; ++i) {
      bf16x16 a = *(const bf16x16*)(A + (int64_t)(img0 + i) * aStride + aoff);
      acc[i] = wmma_bf16(a, b, acc[i]);
    }
  }
#pragma unroll
  for (int i = 0; i < IMGS; ++i) {
    __bf16* Oi = O + (int64_t)(img0 + i) * oStride;
#pragma unroll
    for (int r = 0; r < 8; ++r) {
      int M = trow * 16 + r + ((lane >> 4) << 3);
      int N = tcol * 16 + (lane & 15);
      Oi[pack_off(M, N, outFmt)] = (__bf16)acc[i][r];
    }
  }
}

// ------------- GEMM, A operand shared (basis), 4 images per wave ------------
__global__ __launch_bounds__(256) void gemm_sA(
    const __bf16* __restrict__ A,
    const __bf16* __restrict__ B, int64_t bStride,
    __bf16* __restrict__ O, int64_t oStride, int outFmt) {
  int wave = threadIdx.x >> 5;
  int lane = threadIdx.x & 31;
  int grp  = blockIdx.x >> 5;
  int tile = ((blockIdx.x & 31) << 3) | wave;
  int trow = tile >> 4, tcol = tile & 15;
  int img0 = grp * IMGS;
  f32x8 acc[IMGS] = {};
#pragma unroll
  for (int kt = 0; kt < 8; ++kt) {
    bf16x16 a = *(const bf16x16*)(A + ((trow * 8 + kt) * 512 + lane * 16));
    int boff = (kt * 16 + tcol) * 512 + lane * 16;
#pragma unroll
    for (int i = 0; i < IMGS; ++i) {
      bf16x16 b = *(const bf16x16*)(B + (int64_t)(img0 + i) * bStride + boff);
      acc[i] = wmma_bf16(a, b, acc[i]);
    }
  }
#pragma unroll
  for (int i = 0; i < IMGS; ++i) {
    __bf16* Oi = O + (int64_t)(img0 + i) * oStride;
#pragma unroll
    for (int r = 0; r < 8; ++r) {
      int M = trow * 16 + r + ((lane >> 4) << 3);
      int N = tcol * 16 + (lane & 15);
      Oi[pack_off(M, N, outFmt)] = (__bf16)acc[i][r];
    }
  }
}

// ---- FFT stage 1 fused: Ur = D*Wr and Ui = D*Wi, D loaded once -------------
__global__ __launch_bounds__(256) void gemm_dft1(
    const __bf16* __restrict__ D, int64_t dStride,
    const __bf16* __restrict__ WrB, const __bf16* __restrict__ WiB,
    __bf16* __restrict__ Ur, __bf16* __restrict__ Ui, int64_t oStride) {
  int wave = threadIdx.x >> 5;
  int lane = threadIdx.x & 31;
  int grp  = blockIdx.x >> 5;
  int tile = ((blockIdx.x & 31) << 3) | wave;
  int trow = tile >> 4, tcol = tile & 15;
  int img0 = grp * IMGS;
  f32x8 accR[IMGS] = {}, accI[IMGS] = {};
#pragma unroll
  for (int kt = 0; kt < 8; ++kt) {
    int boff = (kt * 16 + tcol) * 512 + lane * 16;
    bf16x16 bR = *(const bf16x16*)(WrB + boff);
    bf16x16 bI = *(const bf16x16*)(WiB + boff);
    int aoff = (trow * 8 + kt) * 512 + lane * 16;
#pragma unroll
    for (int i = 0; i < IMGS; ++i) {
      bf16x16 a = *(const bf16x16*)(D + (int64_t)(img0 + i) * dStride + aoff);
      accR[i] = wmma_bf16(a, bR, accR[i]);
      accI[i] = wmma_bf16(a, bI, accI[i]);
    }
  }
#pragma unroll
  for (int i = 0; i < IMGS; ++i) {
    __bf16* Or = Ur + (int64_t)(img0 + i) * oStride;
    __bf16* Oi = Ui + (int64_t)(img0 + i) * oStride;
#pragma unroll
    for (int r = 0; r < 8; ++r) {
      int M = trow * 16 + r + ((lane >> 4) << 3);
      int N = tcol * 16 + (lane & 15);
      int off = pack_off(M, N, 0);   // B-pack for FFT stage 2
      Or[off] = (__bf16)accR[i][r];
      Oi[off] = (__bf16)accI[i][r];
    }
  }
}

// ---- FFT stage 2 fused: Fr/Fi in regs -> log|F| -> radial bins -------------
__global__ __launch_bounds__(256) void gemm_mag(
    const __bf16* __restrict__ WrA, const __bf16* __restrict__ WiA,
    const __bf16* __restrict__ Ur, const __bf16* __restrict__ Ui,
    int64_t str, float* __restrict__ tbin) {
  __shared__ float bins[IMGS][192];
  int tid = threadIdx.x;
  if (tid < 192) {
#pragma unroll
    for (int i = 0; i < IMGS; ++i) bins[i][tid] = 0.0f;
  }
  __syncthreads();
  int wave = tid >> 5, lane = tid & 31;
  int grp  = blockIdx.x >> 5;
  int tile = ((blockIdx.x & 31) << 3) | wave;
  int trow = tile >> 4, tcol = tile & 15;
  int img0 = grp * IMGS;
  f32x8 aR1[IMGS] = {}, aR2[IMGS] = {}, aI[IMGS] = {};
#pragma unroll
  for (int kt = 0; kt < 8; ++kt) {
    int aoff = (trow * 8 + kt) * 512 + lane * 16;
    bf16x16 a1 = *(const bf16x16*)(WrA + aoff);   // cos
    bf16x16 a2 = *(const bf16x16*)(WiA + aoff);   // -sin
    int boff = (kt * 16 + tcol) * 512 + lane * 16;
#pragma unroll
    for (int i = 0; i < IMGS; ++i) {
      bf16x16 b1 = *(const bf16x16*)(Ur + (int64_t)(img0 + i) * str + boff);
      bf16x16 b2 = *(const bf16x16*)(Ui + (int64_t)(img0 + i) * str + boff);
      aR1[i] = wmma_bf16(a1, b1, aR1[i]);
      aR2[i] = wmma_bf16(a2, b2, aR2[i]);
      aI[i]  = wmma_bf16(a1, b2, aI[i]);
      aI[i]  = wmma_bf16(a2, b1, aI[i]);
    }
  }
#pragma unroll
  for (int i = 0; i < IMGS; ++i) {
#pragma unroll
    for (int r = 0; r < 8; ++r) {
      float re  = (aR1[i][r] - aR2[i][r]) + 1e-8f;
      float im  = aI[i][r] + 1e-8f;
      float mag = logf(sqrtf(re * re + im * im + 1e-10f) + 1e-10f);
      int M = trow * 16 + r + ((lane >> 4) << 3);
      int N = tcol * 16 + (lane & 15);
      atomicAdd(&bins[i][radial_bin(M, N)], mag);
    }
  }
  __syncthreads();
  if (tid < 192) {
#pragma unroll
    for (int i = 0; i < IMGS; ++i) {
      float v = bins[i][tid];
      if (v != 0.0f) atomicAdd(&tbin[(img0 + i) * 192 + tid], v);
    }
  }
}

// -------------------- profile -> normalize -> linear head -------------------
__global__ __launch_bounds__(256) void final_k(const float* __restrict__ tbin,
                                               const float* __restrict__ nr,
                                               const float* __restrict__ w,
                                               const float* __restrict__ bias,
                                               float* __restrict__ out) {
  int img = blockIdx.x;
  int t   = threadIdx.x;
  __shared__ float prof[192];
  __shared__ float red[256];
  if (t < 182) prof[t] = tbin[img * 192 + t] / (nr[t] + 1e-10f);
  __syncthreads();
  bool valid = (t >= 1) && (t <= 179);  // prof[:,1:-2]
  red[t] = valid ? prof[t] : 3.4e38f;
  __syncthreads();
  for (int s = 128; s > 0; s >>= 1) {
    if (t < s) red[t] = fminf(red[t], red[t + s]);
    __syncthreads();
  }
  float mn = red[0];
  __syncthreads();
  red[t] = valid ? prof[t] : -3.4e38f;
  __syncthreads();
  for (int s = 128; s > 0; s >>= 1) {
    if (t < s) red[t] = fmaxf(red[t], red[t + s]);
    __syncthreads();
  }
  float mx = red[0];
  __syncthreads();
  float denom = mx - mn;
  float contrib = 0.0f;
  if (t >= 90 && t <= 179) {            // last 90 of sliced profile
    float f = (prof[t] - mn) / denom;
    if (f != f) f = 0.0f;               // NaN -> 0
    contrib = f * w[t - 90];
  }
  red[t] = contrib;
  __syncthreads();
  for (int s = 128; s > 0; s >>= 1) {
    if (t < s) red[t] += red[t + s];
    __syncthreads();
  }
  if (t == 0) out[img] = red[0] + bias[0];
}

// ---------------------------------------------------------------------------
extern "C" void kernel_launch(void* const* d_in, const int* in_sizes, int n_in,
                              void* d_out, int out_size, void* d_ws,
                              size_t ws_size, hipStream_t stream) {
  (void)in_sizes; (void)n_in; (void)out_size; (void)ws_size;
  const float* x    = (const float*)d_in[0];  // (128,3,256,256) f32
  const float* w    = (const float*)d_in[1];  // (1,90) f32
  const float* bias = (const float*)d_in[2];  // (1,)  f32
  float* out = (float*)d_out;                 // 128 f32

  char* ws = (char*)d_ws;
  __bf16* basis = (__bf16*)ws;                        // 6*65536 bf16 = 768 KB
  __bf16* CT_B  = basis + 0 * 65536;
  __bf16* C_A   = basis + 1 * 65536;
  __bf16* WR_B  = basis + 2 * 65536;
  __bf16* WI_B  = basis + 3 * 65536;
  __bf16* WR_A  = basis + 4 * 65536;
  __bf16* WI_A  = basis + 5 * 65536;
  float* nr   = (float*)(ws + 786432);                // 192 f32
  float* tbin = (float*)(ws + 787456);                // 128*192 f32
  __bf16* P0 = (__bf16*)(ws + (1u  << 20));           // 16 MB: G -> D
  __bf16* P1 = (__bf16*)(ws + (17u << 20));           // 16 MB: T -> Ur
  __bf16* P2 = (__bf16*)(ws + (33u << 20));           // 16 MB: Ui

  const int64_t IS = 65536;  // per-image packed stride (elements)

  zero_k<<<96, 256, 0, stream>>>(nr, tbin);
  init_basis<<<1536, 256, 0, stream>>>(basis);
  compute_nr<<<256, 256, 0, stream>>>(nr);
  gray_pack<<<8192, 256, 0, stream>>>(x, P0);                      // G (A-pack)
  gemm_sB<<<1024, 256, 0, stream>>>(P0, IS, CT_B, P1, IS, 0);      // T = G*C^T
  gemm_sA<<<1024, 256, 0, stream>>>(C_A, P1, IS, P0, IS, 1);       // D = C*T
  gemm_dft1<<<1024, 256, 0, stream>>>(P0, IS, WR_B, WI_B, P1, P2, IS);
  gemm_mag<<<1024, 256, 0, stream>>>(WR_A, WI_A, P1, P2, IS, tbin);
  final_k<<<128, 256, 0, stream>>>(tbin, nr, w, bias, out);
}
